// DecoderAutoregCrossAttV3_90658169684692
// MI455X (gfx1250) — compile-verified, matrix operand order checked by reading, not run on hardware
//
#include <hip/hip_runtime.h>
#include <cstddef>

typedef _Float16 half_t;
typedef __attribute__((ext_vector_type(16))) _Float16 v16h;
typedef __attribute__((ext_vector_type(8)))  float    v8f;
typedef __attribute__((ext_vector_type(4)))  unsigned int v4u;

// ---------------- geometry ----------------
#define ROWS 96          // B*N = 2*48
#define DDIM 256
#define MDIM 64
#define DFFN 1024
#define AS   264         // LDS A-stride in halves (16B-aligned rows)

// ---------------- workspace layout (bytes) ----------------
static constexpr size_t OFF_BUF  = 0;                       // 96*256 f32
static constexpr size_t OFF_PE   = OFF_BUF  + 98304;        // 48*256 f32
static constexpr size_t OFF_X    = OFF_PE   + 49152;        // 96*256 f32
static constexpr size_t OFF_T    = OFF_X    + 98304;        // 96*256 f32
static constexpr size_t OFF_Q    = OFF_T    + 98304;        // [2][8][48][32] f16
static constexpr size_t OFF_K    = OFF_Q    + 49152;        // [2][8][48][32] f16
static constexpr size_t OFF_VT   = OFF_K    + 49152;        // [2][8][32][64] f16 (s padded to 64, zero tail)
static constexpr size_t OFF_AW   = OFF_VT   + 65536;        // [2][8][48][64] f16 (cols 48..63 zero)
static constexpr size_t OFF_HID  = OFF_AW   + 98304;        // 96*1024 f16  (prep reuses as 2*96*256 f32)
static constexpr size_t OFF_CA   = OFF_HID  + 196608;       // [2][96][256] f32  precomputed cross-attn output
static constexpr size_t OFF_WBF  = OFF_CA   + 196608;       // f16 weights: per layer wq,wk,wv,wo,w1,w2
static constexpr size_t LAYER_HALVES = 65536*4 + 262144*2;  // 786432
static constexpr size_t OFF_LAST = OFF_WBF + 2*LAYER_HALVES*2; // [2][64] f32
static constexpr size_t WS_NEED  = OFF_LAST + 512;

// ---------------- fragment helpers ----------------
union FragA { v16h h; v4u q[2]; };

__device__ inline v16h frag_load(const half_t* base, int row0, int stride, int k0, int lane) {
  // element e of A/B fragment = M[row0 + (lane&15)][k0 + (e>>3)*16 + (lane>>4)*8 + (e&7)]
  const half_t* p = base + (size_t)(row0 + (lane & 15)) * stride + k0 + ((lane >> 4) << 3);
  FragA u;
  u.q[0] = *reinterpret_cast<const v4u*>(p);
  u.q[1] = *reinterpret_cast<const v4u*>(p + 16);
  return u.h;
}

__device__ inline v8f wmma32(v16h a, v16h b, v8f c) {
  return __builtin_amdgcn_wmma_f32_16x16x32_f16(false, a, false, b, (short)0, c, false, false);
}

__device__ inline float allred32(float v) {
  for (int o = 16; o >= 1; o >>= 1) v += __shfl_xor(v, o, 32);
  return v;
}

// ---------------- prep kernels ----------------
__global__ void k_zero(unsigned int* p, int n) {
  for (int i = blockIdx.x * blockDim.x + threadIdx.x; i < n; i += gridDim.x * blockDim.x) p[i] = 0u;
}

__global__ void k_f2h(const float* s, half_t* d, int n) {
  for (int i = blockIdx.x * blockDim.x + threadIdx.x; i < n; i += gridDim.x * blockDim.x)
    d[i] = (half_t)s[i];
}

__global__ void k_pe(float* pe) {
  int t = blockIdx.x, c = threadIdx.x;
  float pos = (float)(t % 30);
  float div = __expf(-(float)(2 * (c >> 1)) * 0.035982976744947866f); // ln(10000)/256
  pe[t * DDIM + c] = (c & 1) ? __cosf(pos * div) : __sinf(pos * div);
}

__global__ void k_cav(const float* content, const float* wv0, const float* bv0,
                      const float* wv1, const float* bv1, float* cav) {
  for (int i = blockIdx.x * blockDim.x + threadIdx.x; i < 2 * ROWS * DDIM; i += gridDim.x * blockDim.x) {
    int l = i / (ROWS * DDIM), rem = i % (ROWS * DDIM), r = rem >> 8, c = rem & 255;
    const float* wv = l ? wv1 : wv0;
    float s = (l ? bv1 : bv0)[c];
    for (int k = 0; k < DDIM; ++k) s += content[r * DDIM + k] * wv[c * DDIM + k];
    cav[i] = s;
  }
}

__global__ void k_cao(const float* cav, const float* wo0, const float* bo0,
                      const float* wo1, const float* bo1, float* cao) {
  for (int i = blockIdx.x * blockDim.x + threadIdx.x; i < 2 * ROWS * DDIM; i += gridDim.x * blockDim.x) {
    int l = i / (ROWS * DDIM), rem = i % (ROWS * DDIM), r = rem >> 8, c = rem & 255;
    const float* wo = l ? wo1 : wo0;
    float s = (l ? bo1 : bo0)[c];
    const float* vrow = cav + (size_t)(l * ROWS + r) * DDIM;
    for (int k = 0; k < DDIM; ++k) s += vrow[k] * wo[c * DDIM + k];
    cao[i] = s;
  }
}

__global__ void k_bufinit(const float* init, const float* style, const float* style_l,
                          const float* Wm, const float* bm, float* buf) {
  int g = blockIdx.x * blockDim.x + threadIdx.x;
  if (g >= 512) return;
  int b = g >> 8, c = g & 255;
  float s = bm[c] + style[b * DDIM + c] + style_l[(size_t)(b * 48) * DDIM + c];
  for (int m = 0; m < MDIM; ++m) s += init[b * MDIM + m] * Wm[c * MDIM + m];
  buf[(size_t)(b * 48) * DDIM + c] = s;
}

// ---------------- main persistent kernel ----------------
struct MainParams {
  const float *style, *style_l;
  const float *sa_bq[2], *sa_bk[2], *sa_bv[2], *sa_bo[2];
  const float *b1[2], *b2[2];
  const float *ln1_g[2], *ln1_b[2], *ln2_g[2], *ln2_b[2], *ln3_g[2], *ln3_b[2];
  const float *mm_b, *mmr_b, *mmr_w, *mm_w;
  char* ws;
  float* out;
};

__global__ __launch_bounds__(512, 1) void k_main(MainParams P) {
  const int tid  = threadIdx.x;
  const int lane = tid & 31;
  const int wave = tid >> 5;            // 0..15
  const int lh   = lane >> 4;
  const int N15  = lane & 15;

  float*  g_buf  = (float*)(P.ws + OFF_BUF);
  float*  g_pe   = (float*)(P.ws + OFF_PE);
  float*  g_x    = (float*)(P.ws + OFF_X);
  float*  g_t    = (float*)(P.ws + OFF_T);
  half_t* g_q    = (half_t*)(P.ws + OFF_Q);
  half_t* g_k    = (half_t*)(P.ws + OFF_K);
  half_t* g_vT   = (half_t*)(P.ws + OFF_VT);
  half_t* g_aw   = (half_t*)(P.ws + OFF_AW);
  half_t* g_hid  = (half_t*)(P.ws + OFF_HID);
  float*  g_ca   = (float*)(P.ws + OFF_CA);
  half_t* wbf    = (half_t*)(P.ws + OFF_WBF);
  float*  g_last = (float*)(P.ws + OFF_LAST);

  __shared__ __align__(16) half_t s_a[ROWS * AS];

  for (int step = 0; step < 48; ++step) {
    // ---- x = buf + pe ----
    for (int idx = tid; idx < ROWS * DDIM; idx += 512) {
      int r = idx >> 8, c = idx & 255;
      float v = g_buf[idx] + g_pe[(r % 48) * DDIM + c];
      g_x[idx] = v;
      s_a[r * AS + c] = (half_t)v;
    }
    __syncthreads();

    for (int l = 0; l < 2; ++l) {
      const half_t* Wq = wbf + (size_t)l * LAYER_HALVES;
      const half_t* Wk = Wq + 65536;
      const half_t* Wv = Wk + 65536;
      const half_t* Wo = Wv + 65536;
      const half_t* W1 = Wo + 65536;
      const half_t* W2 = W1 + 262144;

      // ---- Q,K,V projections: 3 * (6x16) tiles ----
      for (int tt = wave; tt < 288; tt += 16) {
        int which = tt / 96, rem = tt % 96;
        int i = rem % 6, j = rem / 6;
        const half_t* W = (which == 0) ? Wq : (which == 1) ? Wk : Wv;
        const float* bias = (which == 0) ? P.sa_bq[l] : (which == 1) ? P.sa_bk[l] : P.sa_bv[l];
        v8f acc = {0.f, 0.f, 0.f, 0.f, 0.f, 0.f, 0.f, 0.f};
        for (int k0 = 0; k0 < DDIM; k0 += 32) {
          v16h a = frag_load(s_a, i * 16, AS, k0, lane);
          v16h b = frag_load(W, j * 16, DDIM, k0, lane);
          acc = wmma32(a, b, acc);
        }
        int col = j * 16 + N15, h = col >> 5, d = col & 31;
        float bv = bias[col];
        for (int r = 0; r < 8; ++r) {
          int row = i * 16 + r + lh * 8;
          int bidx = row / 48, t = row % 48;
          half_t hv = (half_t)(acc[r] + bv);
          if (which == 0)      g_q[((bidx * 8 + h) * 48 + t) * 32 + d] = hv;
          else if (which == 1) g_k[((bidx * 8 + h) * 48 + t) * 32 + d] = hv;
          else                 g_vT[((bidx * 8 + h) * 32 + d) * 64 + t] = hv; // transposed, s-padded
        }
      }
      __threadfence_block();
      __syncthreads();

      // ---- self-attention: wave w owns (b,h) pair w ----
      {
        int b = wave >> 3, h = wave & 7;
        const half_t* qb = g_q + (size_t)((b * 8 + h) * 48) * 32;
        const half_t* kb = g_k + (size_t)((b * 8 + h) * 48) * 32;
        const half_t* vb = g_vT + (size_t)((b * 8 + h) * 32) * 64;
        half_t* awb = g_aw + (size_t)((b * 8 + h) * 48) * 64;

        v16h af[3], bfr[3];
        for (int i = 0; i < 3; ++i) {
          af[i]  = frag_load(qb, i * 16, 32, 0, lane);
          bfr[i] = frag_load(kb, i * 16, 32, 0, lane);
        }
        v8f sc[3][3];
        for (int i = 0; i < 3; ++i)
          for (int j = 0; j < 3; ++j) {
            v8f z = {0.f, 0.f, 0.f, 0.f, 0.f, 0.f, 0.f, 0.f};
            sc[i][j] = wmma32(af[i], bfr[j], z);
          }
        const float scale = 0.1767766952966369f;        // 1/sqrt(32)
        const float slope = exp2f(-(float)(h + 1));     // ALiBi slope, H=8
        for (int i = 0; i < 3; ++i) {
          for (int r = 0; r < 8; ++r) {
            int t = i * 16 + r + lh * 8;
            float vj[3];
            for (int j = 0; j < 3; ++j) {
              int s = j * 16 + N15;
              float m = (s > t) ? -1e9f : -(float)((t - s) / 30) * slope;
              vj[j] = sc[i][j][r] * scale + m;
            }
            float mx = fmaxf(fmaxf(vj[0], vj[1]), vj[2]);
            for (int o = 8; o >= 1; o >>= 1) mx = fmaxf(mx, __shfl_xor(mx, o, 32)); // 16-lane half
            float e0 = __expf(vj[0] - mx), e1 = __expf(vj[1] - mx), e2 = __expf(vj[2] - mx);
            float sum = e0 + e1 + e2;
            for (int o = 8; o >= 1; o >>= 1) sum += __shfl_xor(sum, o, 32);
            float inv = 1.f / sum;
            awb[t * 64 + N15]      = (half_t)(e0 * inv);
            awb[t * 64 + 16 + N15] = (half_t)(e1 * inv);
            awb[t * 64 + 32 + N15] = (half_t)(e2 * inv);
          }
        }
        asm volatile("s_wait_storecnt 0" ::: "memory"); // same-wave store->load on awb
        for (int i = 0; i < 3; ++i) {
          for (int jd = 0; jd < 2; ++jd) {
            v8f acc = {0.f, 0.f, 0.f, 0.f, 0.f, 0.f, 0.f, 0.f};
            for (int ks = 0; ks < 2; ++ks) {
              v16h a  = frag_load(awb, i * 16, 64, ks * 32, lane);
              v16h bb = frag_load(vb, jd * 16, 64, ks * 32, lane);
              acc = wmma32(a, bb, acc);
            }
            for (int r = 0; r < 8; ++r) {
              int t = i * 16 + r + lh * 8;
              s_a[(b * 48 + t) * AS + h * 32 + jd * 16 + N15] = (half_t)acc[r];
            }
          }
        }
      }
      __threadfence_block();
      __syncthreads();

      // ---- out-proj + residual ----
      for (int tt = wave; tt < 96; tt += 16) {
        int i = tt % 6, j = tt / 6;
        v8f acc = {0.f, 0.f, 0.f, 0.f, 0.f, 0.f, 0.f, 0.f};
        for (int k0 = 0; k0 < DDIM; k0 += 32) {
          v16h a = frag_load(s_a, i * 16, AS, k0, lane);
          v16h b = frag_load(Wo, j * 16, DDIM, k0, lane);
          acc = wmma32(a, b, acc);
        }
        int col = j * 16 + N15;
        float bv = P.sa_bo[l][col];
        for (int r = 0; r < 8; ++r) {
          int row = i * 16 + r + lh * 8;
          g_t[row * DDIM + col] = acc[r] + bv + g_x[row * DDIM + col];
        }
      }
      __threadfence_block();
      __syncthreads();

      // ---- LN1 -> +cross-attn(precomputed) -> LN2 ----
      for (int row = wave; row < ROWS; row += 16) {
        float xv[8], x2[8];
        float sum = 0.f;
        for (int u = 0; u < 8; ++u) { xv[u] = g_t[row * DDIM + lane * 8 + u]; sum += xv[u]; }
        float mean = allred32(sum) * (1.f / 256.f);
        float var = 0.f;
        for (int u = 0; u < 8; ++u) { float d = xv[u] - mean; var += d * d; }
        float rstd = rsqrtf(allred32(var) * (1.f / 256.f) + 1e-5f);
        float sum2 = 0.f;
        for (int u = 0; u < 8; ++u) {
          int c = lane * 8 + u;
          float x1 = (xv[u] - mean) * rstd * P.ln1_g[l][c] + P.ln1_b[l][c];
          x2[u] = x1 + g_ca[((size_t)l * ROWS + row) * DDIM + c];
          sum2 += x2[u];
        }
        float mean2 = allred32(sum2) * (1.f / 256.f);
        float var2 = 0.f;
        for (int u = 0; u < 8; ++u) { float d = x2[u] - mean2; var2 += d * d; }
        float rstd2 = rsqrtf(allred32(var2) * (1.f / 256.f) + 1e-5f);
        for (int u = 0; u < 8; ++u) {
          int c = lane * 8 + u;
          float xo = (x2[u] - mean2) * rstd2 * P.ln2_g[l][c] + P.ln2_b[l][c];
          g_x[row * DDIM + c] = xo;
          s_a[row * AS + c] = (half_t)xo;
        }
      }
      __threadfence_block();
      __syncthreads();

      // ---- FFN1 (96x1024), relu ----
      for (int tt = wave; tt < 384; tt += 16) {
        int i = tt % 6, j = tt / 6;
        v8f acc = {0.f, 0.f, 0.f, 0.f, 0.f, 0.f, 0.f, 0.f};
        for (int k0 = 0; k0 < DDIM; k0 += 32) {
          v16h a = frag_load(s_a, i * 16, AS, k0, lane);
          v16h b = frag_load(W1, j * 16, DDIM, k0, lane);
          acc = wmma32(a, b, acc);
        }
        int col = j * 16 + N15;
        float bv = P.b1[l][col];
        for (int r = 0; r < 8; ++r) {
          int row = i * 16 + r + lh * 8;
          float hv = acc[r] + bv;
          g_hid[(size_t)row * DFFN + col] = (half_t)(hv > 0.f ? hv : 0.f);
        }
      }
      __threadfence_block();
      __syncthreads();

      // ---- FFN2 (96x256, K=1024) + residual ----
      for (int tt = wave; tt < 96; tt += 16) {
        int i = tt % 6, j = tt / 6;
        v8f acc = {0.f, 0.f, 0.f, 0.f, 0.f, 0.f, 0.f, 0.f};
        for (int k0 = 0; k0 < DFFN; k0 += 32) {
          __builtin_prefetch(W2 + (size_t)(j * 16) * DFFN + k0 + 64, 0, 1);
          v16h a = frag_load(g_hid, i * 16, DFFN, k0, lane);
          v16h b = frag_load(W2, j * 16, DFFN, k0, lane);
          acc = wmma32(a, b, acc);
        }
        int col = j * 16 + N15;
        float bv = P.b2[l][col];
        for (int r = 0; r < 8; ++r) {
          int row = i * 16 + r + lh * 8;
          g_t[row * DDIM + col] = acc[r] + bv + g_x[row * DDIM + col];
        }
      }
      __threadfence_block();
      __syncthreads();

      // ---- LN3 ----
      for (int row = wave; row < ROWS; row += 16) {
        float xv[8];
        float sum = 0.f;
        for (int u = 0; u < 8; ++u) { xv[u] = g_t[row * DDIM + lane * 8 + u]; sum += xv[u]; }
        float mean = allred32(sum) * (1.f / 256.f);
        float var = 0.f;
        for (int u = 0; u < 8; ++u) { float d = xv[u] - mean; var += d * d; }
        float rstd = rsqrtf(allred32(var) * (1.f / 256.f) + 1e-5f);
        for (int u = 0; u < 8; ++u) {
          int c = lane * 8 + u;
          float xo = (xv[u] - mean) * rstd * P.ln3_g[l][c] + P.ln3_b[l][c];
          g_x[row * DDIM + c] = xo;
          s_a[row * AS + c] = (half_t)xo;
        }
      }
      __threadfence_block();
      __syncthreads();
    } // layers

    if (step == 47) {
      // final dec_out = x @ Wr.T + br -> d_out (f32, exact weights)
      for (int idx = tid; idx < ROWS * MDIM; idx += 512) {
        int r = idx >> 6, o = idx & 63;
        float s = P.mmr_b[o];
        for (int k2 = 0; k2 < DDIM; ++k2) s += g_x[r * DDIM + k2] * P.mmr_w[o * DDIM + k2];
        P.out[idx] = s;
      }
    } else {
      // last = dec_out[:, step]; new = last @ Wm.T + bm + style + style_l[:, step]
      if (tid < 128) {
        int b = tid >> 6, o = tid & 63;
        int row = b * 48 + step;
        float s = P.mmr_b[o];
        for (int k2 = 0; k2 < DDIM; ++k2) s += g_x[row * DDIM + k2] * P.mmr_w[o * DDIM + k2];
        g_last[b * MDIM + o] = s;
      }
      __threadfence_block();
      __syncthreads();
      {
        int b = tid >> 8, c = tid & 255;
        float s = P.mm_b[c] + P.style[b * DDIM + c] + P.style_l[(size_t)(b * 48 + step) * DDIM + c];
        for (int m = 0; m < MDIM; ++m) s += g_last[b * MDIM + m] * P.mm_w[c * MDIM + m];
        g_buf[(size_t)(b * 48 + step + 1) * DDIM + c] = s;
      }
      __threadfence_block();
      __syncthreads();
    }
  } // steps
}

// ---------------- host launcher ----------------
// params flattening (jax tree order: sorted dict keys, list order):
//   d_in[0..3] = content_code, style_code, style_code_l, init_state
//   per-layer leaves (26 each), layer l at 4 + 26*l:
//   0 b1, 1 b2, 2 ca_bk, 3 ca_bo, 4 ca_bq, 5 ca_bv, 6 ca_wk, 7 ca_wo, 8 ca_wq, 9 ca_wv,
//   10 ln1_b, 11 ln1_g, 12 ln2_b, 13 ln2_g, 14 ln3_b, 15 ln3_g,
//   16 sa_bk, 17 sa_bo, 18 sa_bq, 19 sa_bv, 20 sa_wk, 21 sa_wo, 22 sa_wq, 23 sa_wv, 24 w1, 25 w2
//   d_in[56] motion_map_b, [57] motion_map_r_b, [58] motion_map_r_w, [59] motion_map_w
extern "C" void kernel_launch(void* const* d_in, const int* in_sizes, int n_in,
                              void* d_out, int out_size, void* d_ws, size_t ws_size,
                              hipStream_t stream) {
  (void)in_sizes; (void)n_in; (void)out_size; (void)ws_size;
  char* ws = (char*)d_ws;
  auto leaf = [&](int l, int j) { return (const float*)d_in[4 + l * 26 + j]; };

  k_zero<<<2048, 256, 0, stream>>>((unsigned int*)ws, (int)(WS_NEED / 4));

  for (int l = 0; l < 2; ++l) {
    half_t* wl = (half_t*)(ws + OFF_WBF) + (size_t)l * LAYER_HALVES;
    k_f2h<<<256, 256, 0, stream>>>(leaf(l, 22), wl,           65536);   // sa_wq
    k_f2h<<<256, 256, 0, stream>>>(leaf(l, 20), wl + 65536,   65536);   // sa_wk
    k_f2h<<<256, 256, 0, stream>>>(leaf(l, 23), wl + 131072,  65536);   // sa_wv
    k_f2h<<<256, 256, 0, stream>>>(leaf(l, 21), wl + 196608,  65536);   // sa_wo
    k_f2h<<<1024, 256, 0, stream>>>(leaf(l, 24), wl + 262144, 262144);  // w1
    k_f2h<<<1024, 256, 0, stream>>>(leaf(l, 25), wl + 524288, 262144);  // w2
  }

  k_pe<<<48, 256, 0, stream>>>((float*)(ws + OFF_PE));

  const float* content = (const float*)d_in[0];
  k_cav<<<192, 256, 0, stream>>>(content, leaf(0, 9), leaf(0, 5), leaf(1, 9), leaf(1, 5),
                                 (float*)(ws + OFF_HID));
  k_cao<<<192, 256, 0, stream>>>((float*)(ws + OFF_HID), leaf(0, 7), leaf(0, 3),
                                 leaf(1, 7), leaf(1, 3), (float*)(ws + OFF_CA));

  k_bufinit<<<2, 256, 0, stream>>>((const float*)d_in[3], (const float*)d_in[1],
                                   (const float*)d_in[2], (const float*)d_in[59],
                                   (const float*)d_in[56], (float*)(ws + OFF_BUF));

  MainParams P;
  P.style   = (const float*)d_in[1];
  P.style_l = (const float*)d_in[2];
  for (int l = 0; l < 2; ++l) {
    P.sa_bq[l] = leaf(l, 18); P.sa_bk[l] = leaf(l, 16);
    P.sa_bv[l] = leaf(l, 19); P.sa_bo[l] = leaf(l, 17);
    P.b1[l] = leaf(l, 0);  P.b2[l] = leaf(l, 1);
    P.ln1_g[l] = leaf(l, 11); P.ln1_b[l] = leaf(l, 10);
    P.ln2_g[l] = leaf(l, 13); P.ln2_b[l] = leaf(l, 12);
    P.ln3_g[l] = leaf(l, 15); P.ln3_b[l] = leaf(l, 14);
  }
  P.mm_b  = (const float*)d_in[56];
  P.mmr_b = (const float*)d_in[57];
  P.mmr_w = (const float*)d_in[58];
  P.mm_w  = (const float*)d_in[59];
  P.ws = ws;
  P.out = (float*)d_out;

  k_main<<<1, 512, 0, stream>>>(P);
}
